// AttentionEGNNConv_42511586296499
// MI455X (gfx1250) — compile-verified, matrix operand order checked by reading, not run on hardware
//
#include <hip/hip_runtime.h>
#include <hip/hip_bf16.h>
#include <math.h>

typedef __attribute__((ext_vector_type(16))) _Float16 v16h;
typedef __attribute__((ext_vector_type(8)))  _Float16 v8h;
typedef __attribute__((ext_vector_type(4)))  _Float16 v4h;
typedef __attribute__((ext_vector_type(8)))  float    v8f;

#define WAVES 4
#define BLOCK 128

// ---------------- helpers ----------------

__device__ __forceinline__ float silu_f(float x) {
    return x / (1.0f + __expf(-x));
}

// order-preserving float<->uint encoding for atomicMax-based segment max
__device__ __forceinline__ unsigned enc_f(float f) {
    unsigned u = __float_as_uint(f);
    return (u & 0x80000000u) ? ~u : (u | 0x80000000u);
}
__device__ __forceinline__ float dec_f(unsigned u) {
    unsigned b = (u & 0x80000000u) ? (u & 0x7FFFFFFFu) : ~u;
    return __uint_as_float(b);
}
#define ENC_NEG_INF 0x007FFFFFu  // enc_f(-inf)

// Fragment loader for v_wmma_f32_16x16x32_f16 (A side, and B side when the
// weight matrix is staged TRANSPOSED, i.e. [col][K] row-major per column).
// Lane (row = lane&15, hi = lane>>4) holds K = {kb..kb+7, kb+16..kb+23},
// kb = kc*32 + hi*8.  p = &lds[line*KP + kb] -> two contiguous 16B runs
// -> two ds_load_b128.
__device__ __forceinline__ v16h load_frag(const _Float16* p) {
    v8h lo = *(const v8h*)(p);
    v8h hi = *(const v8h*)(p + 16);
    return __builtin_shufflevector(lo, hi, 0, 1, 2, 3, 4, 5, 6, 7,
                                   8, 9, 10, 11, 12, 13, 14, 15);
}

__device__ __forceinline__ v8f wmma32(v16h a, v16h b, v8f c) {
    return __builtin_amdgcn_wmma_f32_16x16x32_f16(
        /*neg_a=*/false, a, /*neg_b=*/false, b,
        /*c_mod=*/(short)0, c, /*reuse_a=*/false, /*reuse_b=*/false);
}

// float4 global load -> 4x f16 packed LDS store (8B aligned)
__device__ __forceinline__ void cvt_store4(_Float16* dstp, const float* srcp) {
    const float4 v = *(const float4*)srcp;
    v4h h;
    h[0] = (_Float16)v.x; h[1] = (_Float16)v.y;
    h[2] = (_Float16)v.z; h[3] = (_Float16)v.w;
    *(v4h*)dstp = h;
}
__device__ __forceinline__ void zero_store4(_Float16* dstp) {
    v4h z = {};
    *(v4h*)dstp = z;
}

// ---------------- init scratch ----------------

__global__ __launch_bounds__(256) void init_ws_kernel(unsigned* e_max, float* denom,
                                                      float* h_neigh, int N) {
    int stride = gridDim.x * blockDim.x;
    int i0 = blockIdx.x * blockDim.x + threadIdx.x;
    for (int i = i0; i < N * 64; i += stride) h_neigh[i] = 0.0f;
    for (int i = i0; i < N; i += stride) { denom[i] = 0.0f; e_max[i] = ENC_NEG_INF; }
}

// ---------------- pass 1: attention scores + segment max ----------------
// Staged A row (KP=160): [h_src 0..63 | ef 64..79 | dist 80 | pad 81..95 | h_dst 96..159]
// W_k is staged transposed WITH the matching row permutation, W_q transposed.

__global__ __launch_bounds__(BLOCK) void edge_score_kernel(
    const float* __restrict__ node_feat, const float* __restrict__ coord,
    const float* __restrict__ edge_feat,
    const float* __restrict__ W_q, const float* __restrict__ W_k,
    const int* __restrict__ src, const int* __restrict__ dst,
    float* __restrict__ e_arr, unsigned* __restrict__ e_max, int E) {

    __shared__ _Float16 sWkT[64 * 96];   // [col][K=96]
    __shared__ _Float16 sWqT[64 * 64];   // [col][K=64]
    __shared__ _Float16 sA[WAVES][16 * 160];

    for (int i = threadIdx.x; i < 64 * 96; i += BLOCK) {
        int n = i / 96, k = i % 96;
        int kr = (k < 64) ? k : (k < 80 ? (65 + (k - 64)) : (k == 80 ? 64 : -1));
        sWkT[i] = (kr >= 0) ? (_Float16)W_k[kr * 64 + n] : (_Float16)0.0f;
    }
    for (int i = threadIdx.x; i < 64 * 64; i += BLOCK) {
        int n = i >> 6, k = i & 63;
        sWqT[i] = (_Float16)W_q[k * 64 + n];
    }

    const int wave = threadIdx.x >> 5, lane = threadIdx.x & 31;
    const int row = lane & 15, hi = lane >> 4;
    const int numTiles = (E + 15) >> 4;
    const int tile = blockIdx.x * WAVES + wave;
    const bool validWave = tile < numTiles;

    int e = validWave ? (tile * 16 + row) : 0;
    if (e >= E) e = 0;
    const int s = src[e], d = dst[e];

    __syncthreads();

    _Float16* Ar = &sA[wave][row * 160];
    if (hi == 0) {
#pragma unroll
        for (int c = 0; c < 64; c += 4) cvt_store4(&Ar[c], &node_feat[s * 64 + c]);
#pragma unroll
        for (int c = 0; c < 32; c += 4) cvt_store4(&Ar[96 + c], &node_feat[d * 64 + c]);
    } else {
#pragma unroll
        for (int c = 32; c < 64; c += 4) cvt_store4(&Ar[96 + c], &node_feat[d * 64 + c]);
#pragma unroll
        for (int c = 0; c < 16; c += 4) cvt_store4(&Ar[64 + c], &edge_feat[e * 16 + c]);
#pragma unroll
        for (int c = 80; c < 96; c += 4) zero_store4(&Ar[c]);
        float dx = coord[s * 3 + 0] - coord[d * 3 + 0];
        float dy = coord[s * 3 + 1] - coord[d * 3 + 1];
        float dz = coord[s * 3 + 2] - coord[d * 3 + 2];
        Ar[80] = (_Float16)sqrtf(dx * dx + dy * dy + dz * dz);
    }
    __syncthreads();

    v8f accK[4], accQ[4];
#pragma unroll
    for (int nt = 0; nt < 4; ++nt) { accK[nt] = {}; accQ[nt] = {}; }

#pragma unroll
    for (int kc = 0; kc < 3; ++kc) {   // msg_k: K = 96
        const int kb = kc * 32 + hi * 8;
        v16h a = load_frag(&sA[wave][row * 160 + kb]);
#pragma unroll
        for (int nt = 0; nt < 4; ++nt)
            accK[nt] = wmma32(a, load_frag(&sWkT[(nt * 16 + row) * 96 + kb]), accK[nt]);
    }
#pragma unroll
    for (int kc = 0; kc < 2; ++kc) {   // msg_q: K = 64, A at col offset 96
        const int kb = kc * 32 + hi * 8;
        v16h a = load_frag(&sA[wave][row * 160 + 96 + kb]);
#pragma unroll
        for (int nt = 0; nt < 4; ++nt)
            accQ[nt] = wmma32(a, load_frag(&sWqT[(nt * 16 + row) * 64 + kb]), accQ[nt]);
    }

    // e[m] = sum_n K[m,n]*Q[m,n] / 8 ; lane owns rows m = r + 8*hi, 4 columns
    float p[8];
#pragma unroll
    for (int r = 0; r < 8; ++r) {
        p[r] = 0.0f;
#pragma unroll
        for (int nt = 0; nt < 4; ++nt) p[r] += accK[nt][r] * accQ[nt][r];
    }
#pragma unroll
    for (int msk = 1; msk < 16; msk <<= 1) {
#pragma unroll
        for (int r = 0; r < 8; ++r) p[r] += __shfl_xor(p[r], msk, 32);
    }

    if (validWave && row == 0) {
#pragma unroll
        for (int r = 0; r < 8; ++r) {
            int m = r + 8 * hi;
            int eg = tile * 16 + m;
            if (eg < E) {
                float ev = p[r] * 0.125f;  // 1/sqrt(64)
                e_arr[eg] = ev;
                atomicMax(&e_max[dst[eg]], enc_f(ev));
            }
        }
    }
}

// ---------------- pass 2: edge MLP + unnormalized weighted scatter ----------------
// Staged f row (KP=160): [h_src 0..63 | h_dst 64..127 | ef 128..143 | dist 144 | pad..159]
// W_e1 staged transposed with matching row permutation, W_e2 transposed.

__global__ __launch_bounds__(BLOCK) void edge_msg_kernel(
    const float* __restrict__ node_feat, const float* __restrict__ coord,
    const float* __restrict__ edge_feat,
    const float* __restrict__ W_e1, const float* __restrict__ b_e1,
    const float* __restrict__ W_e2, const float* __restrict__ b_e2,
    const int* __restrict__ src, const int* __restrict__ dst,
    const float* __restrict__ e_arr, const unsigned* __restrict__ e_max,
    float* __restrict__ denom, float* __restrict__ h_neigh, int E) {

    __shared__ _Float16 sW1T[64 * 160];  // [col][K=160]
    __shared__ _Float16 sW2T[64 * 64];   // [col][K=64]
    __shared__ _Float16 sA[WAVES][16 * 160];
    __shared__ _Float16 sMid[WAVES][16 * 64];
    __shared__ float    sW16[WAVES][16];
    __shared__ int      sD16[WAVES][16];

    for (int i = threadIdx.x; i < 64 * 160; i += BLOCK) {
        int n = i / 160, k = i % 160;
        int kr = (k < 128) ? k : (k < 144 ? (129 + (k - 128)) : (k == 144 ? 128 : -1));
        sW1T[i] = (kr >= 0) ? (_Float16)W_e1[kr * 64 + n] : (_Float16)0.0f;
    }
    for (int i = threadIdx.x; i < 64 * 64; i += BLOCK) {
        int n = i >> 6, k = i & 63;
        sW2T[i] = (_Float16)W_e2[k * 64 + n];
    }

    const int wave = threadIdx.x >> 5, lane = threadIdx.x & 31;
    const int row = lane & 15, hi = lane >> 4;
    const int numTiles = (E + 15) >> 4;
    const int tile = blockIdx.x * WAVES + wave;
    const bool validWave = tile < numTiles;

    int e = validWave ? (tile * 16 + row) : 0;
    if (e >= E) e = 0;
    const int s = src[e], d = dst[e];

    __syncthreads();

    _Float16* Ar = &sA[wave][row * 160];
    if (hi == 0) {
#pragma unroll
        for (int c = 0; c < 64; c += 4) cvt_store4(&Ar[c], &node_feat[s * 64 + c]);
#pragma unroll
        for (int c = 0; c < 32; c += 4) cvt_store4(&Ar[64 + c], &node_feat[d * 64 + c]);
    } else {
#pragma unroll
        for (int c = 32; c < 64; c += 4) cvt_store4(&Ar[64 + c], &node_feat[d * 64 + c]);
#pragma unroll
        for (int c = 0; c < 16; c += 4) cvt_store4(&Ar[128 + c], &edge_feat[e * 16 + c]);
#pragma unroll
        for (int c = 144; c < 160; c += 4) zero_store4(&Ar[c]);
        float dx = coord[s * 3 + 0] - coord[d * 3 + 0];
        float dy = coord[s * 3 + 1] - coord[d * 3 + 1];
        float dz = coord[s * 3 + 2] - coord[d * 3 + 2];
        Ar[144] = (_Float16)sqrtf(dx * dx + dy * dy + dz * dz);
    }
    __syncthreads();

    // layer 1: [16x160] x [160x64]
    v8f acc[4];
#pragma unroll
    for (int nt = 0; nt < 4; ++nt) acc[nt] = {};
#pragma unroll
    for (int kc = 0; kc < 5; ++kc) {
        const int kb = kc * 32 + hi * 8;
        v16h a = load_frag(&sA[wave][row * 160 + kb]);
#pragma unroll
        for (int nt = 0; nt < 4; ++nt)
            acc[nt] = wmma32(a, load_frag(&sW1T[(nt * 16 + row) * 160 + kb]), acc[nt]);
    }
    // bias + silu -> mid (A-layout, row-major [m][64] so layer-2 A loads are b128)
#pragma unroll
    for (int nt = 0; nt < 4; ++nt) {
        float bb = b_e1[nt * 16 + row];
#pragma unroll
        for (int r = 0; r < 8; ++r) {
            float h = silu_f(acc[nt][r] + bb);
            sMid[wave][(r + 8 * hi) * 64 + nt * 16 + row] = (_Float16)h;
        }
    }
    // softmax weight w = exp(e - max) and denom accumulation (one lane per edge)
    if (hi == 0) {
        int eg = validWave ? (tile * 16 + row) : 0;
        if (eg >= E) eg = 0;
        int dd = dst[eg];
        float w = __expf(e_arr[eg] - dec_f(e_max[dd]));
        sW16[wave][row] = w;
        sD16[wave][row] = dd;
        if (validWave && (tile * 16 + row) < E) atomicAdd(&denom[dd], w);
    }
    __syncthreads();

    // layer 2: [16x64] x [64x64]
    v8f acc2[4];
#pragma unroll
    for (int nt = 0; nt < 4; ++nt) acc2[nt] = {};
#pragma unroll
    for (int kc = 0; kc < 2; ++kc) {
        const int kb = kc * 32 + hi * 8;
        v16h a = load_frag(&sMid[wave][row * 64 + kb]);
#pragma unroll
        for (int nt = 0; nt < 4; ++nt)
            acc2[nt] = wmma32(a, load_frag(&sW2T[(nt * 16 + row) * 64 + kb]), acc2[nt]);
    }
    if (validWave) {
#pragma unroll
        for (int nt = 0; nt < 4; ++nt) {
            float bb = b_e2[nt * 16 + row];
#pragma unroll
            for (int r = 0; r < 8; ++r) {
                int m = r + 8 * hi;
                if (tile * 16 + m < E) {
                    float msg = silu_f(acc2[nt][r] + bb);
                    atomicAdd(&h_neigh[(long)sD16[wave][m] * 64 + nt * 16 + row],
                              msg * sW16[wave][m]);
                }
            }
        }
    }
}

// ---------------- pass 3: node MLP ----------------
// A = [ node_feat(64) | h_neigh/denom(64) ] -> W_n1 -> silu -> W_n2 + b_n2

__global__ __launch_bounds__(BLOCK) void node_mlp_kernel(
    const float* __restrict__ node_feat,
    const float* __restrict__ W_n1, const float* __restrict__ b_n1,
    const float* __restrict__ W_n2, const float* __restrict__ b_n2,
    const float* __restrict__ denom, const float* __restrict__ h_neigh,
    float* __restrict__ out, int N) {

    __shared__ _Float16 sW1T[64 * 128];  // [col][K=128]
    __shared__ _Float16 sW2T[64 * 64];   // [col][K=64]
    __shared__ _Float16 sA[WAVES][16 * 128];
    __shared__ _Float16 sMid[WAVES][16 * 64];

    for (int i = threadIdx.x; i < 64 * 128; i += BLOCK) {
        int n = i >> 7, k = i & 127;
        sW1T[i] = (_Float16)W_n1[k * 64 + n];
    }
    for (int i = threadIdx.x; i < 64 * 64; i += BLOCK) {
        int n = i >> 6, k = i & 63;
        sW2T[i] = (_Float16)W_n2[k * 64 + n];
    }

    const int wave = threadIdx.x >> 5, lane = threadIdx.x & 31;
    const int row = lane & 15, hi = lane >> 4;
    const int numTiles = (N + 15) >> 4;
    const int tile = blockIdx.x * WAVES + wave;
    const bool validWave = tile < numTiles;

    int node = validWave ? (tile * 16 + row) : 0;
    if (node >= N) node = 0;

    __syncthreads();

    _Float16* Ar = &sA[wave][row * 128];
    if (hi == 0) {
#pragma unroll
        for (int c = 0; c < 64; c += 4) cvt_store4(&Ar[c], &node_feat[node * 64 + c]);
    } else {
        float dn = denom[node];
        float inv = (dn > 0.0f) ? (1.0f / dn) : 0.0f;
#pragma unroll
        for (int c = 0; c < 64; c += 4) {
            const float4 v = *(const float4*)&h_neigh[node * 64 + c];
            v4h h;
            h[0] = (_Float16)(v.x * inv); h[1] = (_Float16)(v.y * inv);
            h[2] = (_Float16)(v.z * inv); h[3] = (_Float16)(v.w * inv);
            *(v4h*)&Ar[64 + c] = h;
        }
    }
    __syncthreads();

    v8f acc[4];
#pragma unroll
    for (int nt = 0; nt < 4; ++nt) acc[nt] = {};
#pragma unroll
    for (int kc = 0; kc < 4; ++kc) {
        const int kb = kc * 32 + hi * 8;
        v16h a = load_frag(&sA[wave][row * 128 + kb]);
#pragma unroll
        for (int nt = 0; nt < 4; ++nt)
            acc[nt] = wmma32(a, load_frag(&sW1T[(nt * 16 + row) * 128 + kb]), acc[nt]);
    }
#pragma unroll
    for (int nt = 0; nt < 4; ++nt) {
        float bb = b_n1[nt * 16 + row];
#pragma unroll
        for (int r = 0; r < 8; ++r) {
            float h = silu_f(acc[nt][r] + bb);
            sMid[wave][(r + 8 * hi) * 64 + nt * 16 + row] = (_Float16)h;
        }
    }
    __syncthreads();

    v8f acc2[4];
#pragma unroll
    for (int nt = 0; nt < 4; ++nt) acc2[nt] = {};
#pragma unroll
    for (int kc = 0; kc < 2; ++kc) {
        const int kb = kc * 32 + hi * 8;
        v16h a = load_frag(&sMid[wave][row * 64 + kb]);
#pragma unroll
        for (int nt = 0; nt < 4; ++nt)
            acc2[nt] = wmma32(a, load_frag(&sW2T[(nt * 16 + row) * 64 + kb]), acc2[nt]);
    }
    if (validWave) {
#pragma unroll
        for (int nt = 0; nt < 4; ++nt) {
            float bb = b_n2[nt * 16 + row];
#pragma unroll
            for (int r = 0; r < 8; ++r) {
                int m = r + 8 * hi;
                int nd = tile * 16 + m;
                if (nd < N) out[(long)nd * 64 + nt * 16 + row] = acc2[nt][r] + bb;
            }
        }
    }
}

// ---------------- launcher ----------------

extern "C" void kernel_launch(void* const* d_in, const int* in_sizes, int n_in,
                              void* d_out, int out_size, void* d_ws, size_t ws_size,
                              hipStream_t stream) {
    const float* node_feat = (const float*)d_in[0];
    const float* coord     = (const float*)d_in[1];
    const float* edge_feat = (const float*)d_in[2];
    const float* W_e1 = (const float*)d_in[3];
    const float* b_e1 = (const float*)d_in[4];
    const float* W_e2 = (const float*)d_in[5];
    const float* b_e2 = (const float*)d_in[6];
    const float* W_n1 = (const float*)d_in[7];
    const float* b_n1 = (const float*)d_in[8];
    const float* W_n2 = (const float*)d_in[9];
    const float* b_n2 = (const float*)d_in[10];
    const float* W_q  = (const float*)d_in[11];
    const float* W_k  = (const float*)d_in[12];
    const int*   src  = (const int*)d_in[13];
    const int*   dst  = (const int*)d_in[14];

    const int N = in_sizes[0] / 64;
    const int E = in_sizes[13];

    // workspace layout: e_arr[E] | e_max[N] | denom[N] | h_neigh[N*64]
    float*    e_arr   = (float*)d_ws;
    unsigned* e_max   = (unsigned*)(e_arr + E);
    float*    denom   = (float*)(e_max + N);
    float*    h_neigh = denom + N;

    int initBlocks = (N * 64 + 255) / 256;
    if (initBlocks > 1024) initBlocks = 1024;
    init_ws_kernel<<<initBlocks, 256, 0, stream>>>(e_max, denom, h_neigh, N);

    const int numEdgeTiles = (E + 15) / 16;
    const int edgeBlocks   = (numEdgeTiles + WAVES - 1) / WAVES;
    const int numNodeTiles = (N + 15) / 16;
    const int nodeBlocks   = (numNodeTiles + WAVES - 1) / WAVES;

    edge_score_kernel<<<edgeBlocks, BLOCK, 0, stream>>>(
        node_feat, coord, edge_feat, W_q, W_k, src, dst, e_arr, e_max, E);

    edge_msg_kernel<<<edgeBlocks, BLOCK, 0, stream>>>(
        node_feat, coord, edge_feat, W_e1, b_e1, W_e2, b_e2,
        src, dst, e_arr, e_max, denom, h_neigh, E);

    node_mlp_kernel<<<nodeBlocks, BLOCK, 0, stream>>>(
        node_feat, W_n1, b_n1, W_n2, b_n2, denom, h_neigh, (float*)d_out, N);
}
